// HAMABlock_70806830842485
// MI455X (gfx1250) — matrix operand order repaired and checked by reading, not compiled, not run on hardware
//
#include <hip/hip_runtime.h>
#include <hip/hip_bf16.h>
#include <math.h>

// ---------------- problem constants ----------------
#define B_ 4
#define L_ 4096
#define D_ 256
#define N_ 64
#define P_ 128
#define NH_ 4
#define DH_ 64
#define FF_ 2048
#define NMASK_ 16
#define STEP_ 62            // max(1,(L-P)//(N-1)) = 3968/63 = 62
#define BN_ (B_ * N_)       // 256 nodes
#define T_ (BN_ * P_)       // 32768 token rows

typedef __attribute__((ext_vector_type(16))) __bf16 v16bf;
typedef __attribute__((ext_vector_type(8)))  float  v8f;

union U128 { uint4 v; unsigned short u[8]; __bf16 h[8]; };

// optional CDNA5 async global->LDS copies (guarded; sync fallback)
#if defined(__has_builtin)
#if __has_builtin(__builtin_amdgcn_global_load_async_to_lds_b128) && \
    __has_builtin(__builtin_amdgcn_s_wait_asynccnt)
#define HAVE_ASYNC_LDS 1
#endif
#endif
#ifndef HAVE_ASYNC_LDS
#define HAVE_ASYNC_LDS 0
#endif

#if HAVE_ASYNC_LDS
// builtin signature (from compiler diagnostic): v4i AS1* src, v4i AS3* dst, Imm, Imm
typedef int async_v4i __attribute__((ext_vector_type(4)));
typedef __attribute__((address_space(1))) async_v4i* gptr_v4i;
typedef __attribute__((address_space(3))) async_v4i* lptr_v4i;
#endif

__device__ __forceinline__ unsigned short f2bf(float f) {
  unsigned int u = __float_as_uint(f);
  u += 0x7FFFu + ((u >> 16) & 1u);   // round-to-nearest-even
  return (unsigned short)(u >> 16);
}

__device__ __forceinline__ v8f wmma_bf16(v16bf a, v16bf b, v8f c) {
  // D(f32 16x16) = A(bf16 16x32) * B(bf16 32x16) + C
  return __builtin_amdgcn_wmma_f32_16x16x32_bf16(false, a, false, b,
                                                 (short)0, c, false, false);
}

// A fragment 16x32 (MxK): src row-major, ld in elements.
// ISA layout: lanes 0-15 row M=lane, K 0..7 / 16..23 ; lanes 16-31 K 8..15 / 24..31
__device__ __forceinline__ v16bf load_a16x32(const unsigned short* src, int ld) {
  int lane = threadIdx.x & 31;
  int r    = lane & 15;
  int kb   = (lane >> 4) << 3;                 // 0 or 8
  const unsigned short* p = src + r * ld + kb;
  U128 lo, hi;
  lo.v = *(const uint4*)(p);
  hi.v = *(const uint4*)(p + 16);
  v16bf a;
#pragma unroll
  for (int i = 0; i < 8; i++) { a[i] = lo.h[i]; a[8 + i] = hi.h[i]; }
  return a;
}

// B fragment 32x16 (KxN): element (k,n) stored at src[n*ld + k] (k contiguous).
// ISA layout: lanes 0-15 K 0..7 / 16..23, lanes 16-31 K 8..15 / 24..31, col N = lane&15.
__device__ __forceinline__ v16bf load_b32x16(const unsigned short* src, int ld) {
  int lane = threadIdx.x & 31;
  int n    = lane & 15;
  int kb   = (lane >> 4) << 3;                 // 0 or 8
  const unsigned short* p = src + n * ld + kb;
  U128 lo, hi;
  lo.v = *(const uint4*)(p);
  hi.v = *(const uint4*)(p + 16);
  v16bf b;
#pragma unroll
  for (int i = 0; i < 8; i++) { b[i] = lo.h[i]; b[8 + i] = hi.h[i]; }
  return b;
}

__device__ __forceinline__ float pe_val(int pos, int d) {
  int dd = d & ~1;
  float ang = (float)pos * __expf((float)dd * (-9.210340371976184f / 256.0f));
  return (d & 1) ? __cosf(ang) : __sinf(ang);
}

// ---------------- weight prep: W[Nc,K] f32 -> Wt[K,Nc] bf16 ----------------
__global__ void prepw_kernel(const float* __restrict__ W,
                             unsigned short* __restrict__ Wt,
                             int Nc, int K, int total) {
  int gid = blockIdx.x * 256 + threadIdx.x;
  if (gid >= total) return;
  int k = gid / Nc, n = gid - k * Nc;
  Wt[gid] = f2bf(W[(size_t)n * K + k]);
}

// ---------------- generic bf16 WMMA GEMM (software-pipelined) ----------------
// C[M,Nc] = A[M,K](bf16,row-major) @ Wt[K,Nc](bf16) + bias (+resid) (relu?)
#define BM 128
#define BN 64
#define BK 32
__global__ void gemm_bf16_kernel(const unsigned short* __restrict__ A,
                                 const unsigned short* __restrict__ Wt,
                                 const float* __restrict__ bias,
                                 const float* __restrict__ resid,
                                 float* __restrict__ outF,
                                 unsigned short* __restrict__ outB,
                                 int M, int Nc, int K, int relu) {
  __shared__ unsigned short As[BM * BK];   // row-major [128][32]
  __shared__ unsigned short Bs[BN * BK];   // n-major  [64][32] (k contiguous)
  int m0   = blockIdx.x * BM;
  int n0   = blockIdx.y * BN;
  int tid  = threadIdx.x;
  int wave = tid >> 5;
  int lane = tid & 31;

  v8f acc[4];
#pragma unroll
  for (int j = 0; j < 4; j++)
#pragma unroll
    for (int i = 0; i < 8; i++) acc[j][i] = 0.0f;

  int arow = tid >> 1, ahalf = tid & 1;
  int bkk  = tid >> 3, bseg  = tid & 7;
  const unsigned short* Aptr = A + (size_t)(m0 + arow) * K + ahalf * 16;
  const unsigned short* Bptr = Wt + (size_t)bkk * Nc + n0 + bseg * 8;
  int nK = K / BK;

  // prologue: first tile into registers
  uint4 ra0 = *(const uint4*)(Aptr);
  uint4 ra1 = *(const uint4*)(Aptr + 8);
  U128  rb;  rb.v = *(const uint4*)(Bptr);

  for (int kt = 0; kt < nK; kt++) {
    __syncthreads();                       // LDS consumers of prev iter done
    {
      uint4* dst = (uint4*)(As + arow * BK + ahalf * 16);
      dst[0] = ra0; dst[1] = ra1;
    }
#pragma unroll
    for (int i = 0; i < 8; i++) Bs[(bseg * 8 + i) * BK + bkk] = rb.u[i];
    __syncthreads();
    if (kt + 1 < nK) {                     // overlap next global fetch w/ WMMAs
      const unsigned short* ap = Aptr + (size_t)(kt + 1) * BK;
      ra0 = *(const uint4*)(ap);
      ra1 = *(const uint4*)(ap + 8);
      rb.v = *(const uint4*)(Bptr + (size_t)(kt + 1) * BK * Nc);
      if (kt + 2 < nK) __builtin_prefetch(Aptr + (size_t)(kt + 2) * BK, 0, 1);
    }
    v16bf a = load_a16x32(As + (wave * 16) * BK, BK);
#pragma unroll
    for (int j = 0; j < 4; j++) {
      v16bf b = load_b32x16(Bs + (j * 16) * BK, BK);
      acc[j] = wmma_bf16(a, b, acc[j]);
    }
  }

  // epilogue: C layout row = i + 8*(lane>=16), col = lane&15
  int roff = (lane >> 4) * 8;
  int c16  = lane & 15;
#pragma unroll
  for (int j = 0; j < 4; j++) {
    int col = n0 + j * 16 + c16;
    float bb = bias ? bias[col] : 0.0f;
#pragma unroll
    for (int i = 0; i < 8; i++) {
      int row = m0 + wave * 16 + roff + i;
      float v = acc[j][i] + bb;
      if (relu) v = fmaxf(v, 0.0f);
      size_t o = (size_t)row * Nc + col;
      if (resid) v += resid[o];
      if (outF) outF[o] = v;
      if (outB) outB[o] = f2bf(v);
    }
  }
}

// ---------------- attention: one block per (node, head) ----------------
// Static LDS (48 KB): Qs[128][64], Ks[128][64], Vt[64][128]; the bf16 prob
// matrix Pb[128][128] aliases Qs+Ks once the S-phase WMMAs are done.
// Softmax is done fully in registers via wave32 shuffles.
__global__ void attn_kernel(const unsigned short* __restrict__ qkv,
                            unsigned short* __restrict__ attout) {
  __shared__ unsigned short QK[2 * 128 * 64];   // Qs | Ks  (aliased by Pb)
  __shared__ unsigned short Vt[64 * 128];       // transposed: [dh][row]
  unsigned short* Qs = QK;
  unsigned short* Ks = QK + 128 * 64;
  unsigned short* Pb = QK;                      // alias (32 KB)

  int bn   = blockIdx.x;
  int head = blockIdx.y;
  int tid  = threadIdx.x;
  int wave = tid >> 5;
  int lane = tid & 31;
  size_t t0 = (size_t)bn * P_;
  int qoff = head * DH_, koff = D_ + head * DH_, voff = 2 * D_ + head * DH_;

  // stage Q,K row-major (async if available); V transposed through registers
#pragma unroll
  for (int rep = 0; rep < 4; rep++) {
    int idx = tid + rep * 256;                  // 0..1023
    int row = idx >> 3, seg = idx & 7;
    const unsigned short* src = qkv + (t0 + row) * (3 * D_);
#if HAVE_ASYNC_LDS
    __builtin_amdgcn_global_load_async_to_lds_b128(
        (gptr_v4i)(src + qoff + seg * 8),
        (lptr_v4i)(Qs + row * 64 + seg * 8), 0, 0);
    __builtin_amdgcn_global_load_async_to_lds_b128(
        (gptr_v4i)(src + koff + seg * 8),
        (lptr_v4i)(Ks + row * 64 + seg * 8), 0, 0);
#else
    *(uint4*)(Qs + row * 64 + seg * 8) = *(const uint4*)(src + qoff + seg * 8);
    *(uint4*)(Ks + row * 64 + seg * 8) = *(const uint4*)(src + koff + seg * 8);
#endif
    U128 rv; rv.v = *(const uint4*)(src + voff + seg * 8);
#pragma unroll
    for (int i = 0; i < 8; i++) Vt[(seg * 8 + i) * 128 + row] = rv.u[i];
  }
#if HAVE_ASYNC_LDS
  __builtin_amdgcn_s_wait_asynccnt(0);
#endif
  __syncthreads();

  int roff = (lane >> 4) * 8;
  int c16  = lane & 15;

  // S = Q @ K^T  (K row-major => B frag (k,n) = Ks[n*64+k], contiguous in k)
  v8f acc[8];
#pragma unroll
  for (int j = 0; j < 8; j++)
#pragma unroll
    for (int i = 0; i < 8; i++) acc[j][i] = 0.0f;
#pragma unroll
  for (int kt = 0; kt < 2; kt++) {
    v16bf a = load_a16x32(Qs + (wave * 16) * 64 + kt * 32, 64);
#pragma unroll
    for (int j = 0; j < 8; j++) {
      v16bf b = load_b32x16(Ks + (j * 16) * 64 + kt * 32, 64);
      acc[j] = wmma_bf16(a, b, acc[j]);
    }
  }
  __syncthreads();   // all waves finished reading Qs/Ks -> Pb alias is safe

  // register softmax: row (16w+roff+i) lives in 16 lanes (same lane>>4 half)
  // across the 8 j-accumulators; shfl_xor masks 1,2,4,8 stay in the half.
#pragma unroll
  for (int i = 0; i < 8; i++) {
    float m = acc[0][i];
#pragma unroll
    for (int j = 1; j < 8; j++) m = fmaxf(m, acc[j][i]);
#pragma unroll
    for (int mk = 1; mk < 16; mk <<= 1) m = fmaxf(m, __shfl_xor(m, mk, 32));
    float sum = 0.0f;
#pragma unroll
    for (int j = 0; j < 8; j++) {
      float e = __expf((acc[j][i] - m) * 0.125f);   // 1/sqrt(DH)=0.125
      acc[j][i] = e; sum += e;
    }
#pragma unroll
    for (int mk = 1; mk < 16; mk <<= 1) sum += __shfl_xor(sum, mk, 32);
    float inv = 1.0f / sum;
    int row = wave * 16 + roff + i;
#pragma unroll
    for (int j = 0; j < 8; j++)
      Pb[row * 128 + j * 16 + c16] = f2bf(acc[j][i] * inv);
  }
  __syncthreads();

  // O = P @ V   (Vt transposed => B frag (k,n) = Vt[n*128+k], contiguous)
  v8f oac[4];
#pragma unroll
  for (int j = 0; j < 4; j++)
#pragma unroll
    for (int i = 0; i < 8; i++) oac[j][i] = 0.0f;
#pragma unroll
  for (int kt = 0; kt < 4; kt++) {
    v16bf a = load_a16x32(Pb + (wave * 16) * 128 + kt * 32, 128);
#pragma unroll
    for (int j = 0; j < 4; j++) {
      v16bf b = load_b32x16(Vt + (j * 16) * 128 + kt * 32, 128);
      oac[j] = wmma_bf16(a, b, oac[j]);
    }
  }
#pragma unroll
  for (int j = 0; j < 4; j++)
#pragma unroll
    for (int i = 0; i < 8; i++) {
      int row = wave * 16 + roff + i;
      attout[(t0 + row) * D_ + head * DH_ + j * 16 + c16] = f2bf(oac[j][i]);
    }
}

// ---------------- LayerNorm over D=256, one row per block (256 thr) -------
__global__ void ln_kernel(const float* __restrict__ in,
                          const float* __restrict__ g,
                          const float* __restrict__ b,
                          float* __restrict__ outF,
                          unsigned short* __restrict__ outB) {
  int row = blockIdx.x, d = threadIdx.x;
  __shared__ float red[256];
  float v = in[(size_t)row * D_ + d];
  red[d] = v; __syncthreads();
  for (int s = 128; s > 0; s >>= 1) { if (d < s) red[d] += red[d + s]; __syncthreads(); }
  float mean = red[0] * (1.0f / D_);
  __syncthreads();
  float c = v - mean;
  red[d] = c * c; __syncthreads();
  for (int s = 128; s > 0; s >>= 1) { if (d < s) red[d] += red[d + s]; __syncthreads(); }
  float var = red[0] * (1.0f / D_);
  float o = c * rsqrtf(var + 1e-5f) * g[d] + b[d];
  size_t idx = (size_t)row * D_ + d;
  if (outF) outF[idx] = o;
  if (outB) outB[idx] = f2bf(o);
}

// ---------------- fission: xp = LN(x)*w+b + PE(L,D) ----------------
__global__ void fiss_kernel(const float* __restrict__ x,
                            const float* __restrict__ g,
                            const float* __restrict__ b,
                            float* __restrict__ xp) {
  int row = blockIdx.x, d = threadIdx.x;
  int l = row & (L_ - 1);
  __shared__ float red[256];
  float v = x[(size_t)row * D_ + d];
  red[d] = v; __syncthreads();
  for (int s = 128; s > 0; s >>= 1) { if (d < s) red[d] += red[d + s]; __syncthreads(); }
  float mean = red[0] * (1.0f / D_);
  __syncthreads();
  float c = v - mean;
  red[d] = c * c; __syncthreads();
  for (int s = 128; s > 0; s >>= 1) { if (d < s) red[d] += red[d + s]; __syncthreads(); }
  float var = red[0] * (1.0f / D_);
  xp[(size_t)row * D_ + d] = c * rsqrtf(var + 1e-5f) * g[d] + b[d] + pe_val(l, d);
}

// ---------------- gather: node rows = xp[b, n*62+p, :] + PE(P,D)[p] -------
__global__ void gather_kernel(const float* __restrict__ xp,
                              float* __restrict__ hF,
                              unsigned short* __restrict__ hB) {
  int t = blockIdx.x, d = threadIdx.x;
  int bn = t >> 7, p = t & 127;
  int b = bn >> 6, n = bn & 63;
  int l = n * STEP_ + p;
  float v = xp[((size_t)(b * L_ + l)) * D_ + d] + pe_val(p, d);
  size_t idx = (size_t)t * D_ + d;
  hF[idx] = v;
  hB[idx] = f2bf(v);
}

// ---------------- mask the NMASK lowest-norm tokens per node ----------------
__global__ void mask_kernel(float* __restrict__ hF, unsigned short* __restrict__ hB) {
  int bn = blockIdx.x, tid = threadIdx.x;      // 128 threads
  __shared__ float sc[128];
  __shared__ int   mk[128];
  size_t t = (size_t)bn * P_ + tid;
  float s = 0.0f;
  for (int d = 0; d < D_; d++) { float v = hF[t * D_ + d]; s += v * v; }
  sc[tid] = s; mk[tid] = 0;
  __syncthreads();
  if (tid == 0) {
    for (int it = 0; it < NMASK_; it++) {
      int am = 0; float mv = sc[0];
      for (int j = 1; j < 128; j++) if (sc[j] < mv) { mv = sc[j]; am = j; }
      mk[am] = 1; sc[am] = 3.0e38f;
    }
  }
  __syncthreads();
  for (int tok = 0; tok < 128; tok++)
    if (mk[tok]) {
      size_t base = ((size_t)bn * P_ + tok) * D_;
      hF[base + tid] = 0.0f; hF[base + tid + 128] = 0.0f;
      hB[base + tid] = 0;    hB[base + tid + 128] = 0;
    }
}

// ---------------- fusion ----------------
__global__ void fill_kernel(float* __restrict__ p, int n) {
  int g = blockIdx.x * 256 + threadIdx.x;
  if (g < n) p[g] = 0.0f;
}

__global__ void scatter_kernel(const float* __restrict__ h, float* __restrict__ acc) {
  int t = blockIdx.x, d = threadIdx.x;
  int bn = t >> 7, p = t & 127;
  int b = bn >> 6, n = bn & 63;
  int l = n * STEP_ + p;
  atomicAdd(&acc[((size_t)(b * L_ + l)) * D_ + d], h[(size_t)t * D_ + d]);
}

__global__ void final_ln_kernel(const float* __restrict__ acc,
                                const float* __restrict__ g,
                                const float* __restrict__ b,
                                float* __restrict__ out) {
  int row = blockIdx.x, d = threadIdx.x;
  int l = row & (L_ - 1);
  int nmin = (l > P_ - 1) ? (l - (P_ - 1) + STEP_ - 1) / STEP_ : 0;
  int nmax = l / STEP_; if (nmax > N_ - 1) nmax = N_ - 1;
  int cnt = nmax - nmin + 1; if (cnt < 1) cnt = 1;
  __shared__ float red[256];
  float v = acc[(size_t)row * D_ + d] / (float)cnt;
  red[d] = v; __syncthreads();
  for (int s = 128; s > 0; s >>= 1) { if (d < s) red[d] += red[d + s]; __syncthreads(); }
  float mean = red[0] * (1.0f / D_);
  __syncthreads();
  float c = v - mean;
  red[d] = c * c; __syncthreads();
  for (int s = 128; s > 0; s >>= 1) { if (d < s) red[d] += red[d + s]; __syncthreads(); }
  float var = red[0] * (1.0f / D_);
  out[(size_t)row * D_ + d] = c * rsqrtf(var + 1e-5f) * g[d] + b[d];
}

// ---------------- host orchestration ----------------
extern "C" void kernel_launch(void* const* d_in, const int* in_sizes, int n_in,
                              void* d_out, int out_size, void* d_ws, size_t ws_size,
                              hipStream_t stream) {
  (void)in_sizes; (void)n_in; (void)out_size; (void)ws_size;

  const float* x      = (const float*)d_in[0];
  const float* fiss_w = (const float*)d_in[1];
  const float* fiss_b = (const float*)d_in[2];
  const float* fus_w  = (const float*)d_in[3];
  const float* fus_b  = (const float*)d_in[4];

  char* base = (char*)d_ws;
  size_t off = 0;
  auto take = [&](size_t bytes) -> void* {
    void* p = base + off;
    off = (off + bytes + 255) & ~(size_t)255;
    return p;
  };
  unsigned short* wtq_e = (unsigned short*)take((size_t)256 * 768 * 2);
  unsigned short* wto_e = (unsigned short*)take((size_t)256 * 256 * 2);
  unsigned short* wt1_e = (unsigned short*)take((size_t)256 * 2048 * 2);
  unsigned short* wt2_e = (unsigned short*)take((size_t)2048 * 256 * 2);
  unsigned short* wtq_d = (unsigned short*)take((size_t)256 * 768 * 2);
  unsigned short* wto_d = (unsigned short*)take((size_t)256 * 256 * 2);
  unsigned short* wt1_d = (unsigned short*)take((size_t)256 * 2048 * 2);
  unsigned short* wt2_d = (unsigned short*)take((size_t)2048 * 256 * 2);
  float*          xp    = (float*)take((size_t)B_ * L_ * D_ * 4);
  float*          h_f   = (float*)take((size_t)T_ * D_ * 4);
  unsigned short* h_b   = (unsigned short*)take((size_t)T_ * D_ * 2);
  unsigned short* qkvb  = (unsigned short*)take((size_t)T_ * 3 * D_ * 2);
  unsigned short* attb  = (unsigned short*)take((size_t)T_ * D_ * 2);
  unsigned short* ffb   = (unsigned short*)take((size_t)T_ * FF_ * 2);
  float*          tmp   = (float*)take((size_t)T_ * D_ * 4);
  float*          accb  = (float*)take((size_t)B_ * L_ * D_ * 4);

  auto prep = [&](const float* W, unsigned short* Wt, int Nc, int K) {
    int tot = Nc * K;
    prepw_kernel<<<(tot + 255) / 256, 256, 0, stream>>>(W, Wt, Nc, K, tot);
  };
  prep((const float*)d_in[5],  wtq_e, 3 * D_, D_);
  prep((const float*)d_in[7],  wto_e, D_,     D_);
  prep((const float*)d_in[13], wt1_e, FF_,    D_);
  prep((const float*)d_in[15], wt2_e, D_,     FF_);
  prep((const float*)d_in[17], wtq_d, 3 * D_, D_);
  prep((const float*)d_in[19], wto_d, D_,     D_);
  prep((const float*)d_in[25], wt1_d, FF_,    D_);
  prep((const float*)d_in[27], wt2_d, D_,     FF_);

  fiss_kernel<<<B_ * L_, 256, 0, stream>>>(x, fiss_w, fiss_b, xp);
  gather_kernel<<<T_, 256, 0, stream>>>(xp, h_f, h_b);

  auto layer = [&](unsigned short* wq, const float* bq,
                   unsigned short* wo, const float* bo,
                   const float* l1w, const float* l1b,
                   const float* l2w, const float* l2b,
                   unsigned short* w1, const float* b1,
                   unsigned short* w2, const float* b2) {
    dim3 gq(T_ / BM, (3 * D_) / BN);
    gemm_bf16_kernel<<<gq, 256, 0, stream>>>(h_b, wq, bq, nullptr, nullptr, qkvb,
                                             T_, 3 * D_, D_, 0);
    dim3 ga(BN_, NH_);
    attn_kernel<<<ga, 256, 0, stream>>>(qkvb, attb);
    dim3 go(T_ / BM, D_ / BN);
    gemm_bf16_kernel<<<go, 256, 0, stream>>>(attb, wo, bo, h_f, tmp, nullptr,
                                             T_, D_, D_, 0);
    ln_kernel<<<T_, 256, 0, stream>>>(tmp, l1w, l1b, h_f, h_b);
    dim3 g1(T_ / BM, FF_ / BN);
    gemm_bf16_kernel<<<g1, 256, 0, stream>>>(h_b, w1, b1, nullptr, nullptr, ffb,
                                             T_, FF_, D_, 1);
    dim3 g2(T_ / BM, D_ / BN);
    gemm_bf16_kernel<<<g2, 256, 0, stream>>>(ffb, w2, b2, h_f, tmp, nullptr,
                                             T_, D_, FF_, 0);
    ln_kernel<<<T_, 256, 0, stream>>>(tmp, l2w, l2b, h_f, h_b);
  };

  // encoder
  layer(wtq_e, (const float*)d_in[6],  wto_e, (const float*)d_in[8],
        (const float*)d_in[9],  (const float*)d_in[10],
        (const float*)d_in[11], (const float*)d_in[12],
        wt1_e, (const float*)d_in[14], wt2_e, (const float*)d_in[16]);

  mask_kernel<<<BN_, 128, 0, stream>>>(h_f, h_b);

  // decoder
  layer(wtq_d, (const float*)d_in[18], wto_d, (const float*)d_in[20],
        (const float*)d_in[21], (const float*)d_in[22],
        (const float*)d_in[23], (const float*)d_in[24],
        wt1_d, (const float*)d_in[26], wt2_d, (const float*)d_in[28]);

  // fusion: scatter-add, average, final LN
  fill_kernel<<<(B_ * L_ * D_) / 256, 256, 0, stream>>>(accb, B_ * L_ * D_);
  scatter_kernel<<<T_, 256, 0, stream>>>(h_f, accb);
  final_ln_kernel<<<B_ * L_, 256, 0, stream>>>(accb, fus_w, fus_b, (float*)d_out);
}
